// MHMoE_Block_40364102648323
// MI455X (gfx1250) — matrix operand order repaired and checked by reading, not compiled, not run on hardware
//
#include <hip/hip_runtime.h>

typedef __attribute__((ext_vector_type(16))) _Float16 v16h;
typedef __attribute__((ext_vector_type(8)))  _Float16 v8h;
typedef __attribute__((ext_vector_type(8)))  float    v8f;

#define ACT_NONE 0
#define ACT_GELU 1
#define ACT_RELU 2
#define ACT_TANH 3

__device__ __forceinline__ float act_apply(float v, int act) {
  if (act == ACT_GELU) return 0.5f * v * (1.f + erff(v * 0.70710678118654752f));
  if (act == ACT_RELU) return v > 0.f ? v : 0.f;
  if (act == ACT_TANH) return tanhf(v);
  return v;
}

__device__ __forceinline__ void zero8(v8h& v) {
  for (int j = 0; j < 8; ++j) v[j] = (_Float16)0.f;
}

// --- CDNA5 async global->LDS copy (ASYNCcnt path, CDNA5 ISA) ---
__device__ __forceinline__ unsigned lds_off(const void* p) {
#if defined(__gfx1250__)
  return (unsigned)(unsigned long long)(__attribute__((address_space(3))) const void*)p;
#else
  return 0;
#endif
}
__device__ __forceinline__ void async_ld_b128(const void* g, unsigned lds) {
#if defined(__gfx1250__)
  asm volatile("global_load_async_to_lds_b128 %0, %1, off"
               :: "v"(lds), "v"(g) : "memory");
#else
  (void)g; (void)lds;
#endif
}
__device__ __forceinline__ void wait_async0() {
#if defined(__gfx1250__)
  asm volatile("s_wait_asynccnt 0x0" ::: "memory");
#endif
}

// Epilogue for one 16x16 D tile (f32 layout: lane n = fr, M = hi*8 + r).
__device__ __forceinline__ void d_epilogue(
    const v8f& acc, int gn, int mbase, int N,
    const float* __restrict__ bias,
    float* __restrict__ C, _Float16* __restrict__ Ch, int ldc,
    const float* __restrict__ Res, int ldr, int act) {
  if (gn >= N) return;
  const float bv = bias ? bias[gn] : 0.f;
  for (int r = 0; r < 8; ++r) {
    const size_t gm = (size_t)(mbase + r);
    float v = acc[r] + bv;
    v = act_apply(v, act);
    if (Res) v += Res[gm * (size_t)ldr + gn];
    if (C)  C[gm * (size_t)ldc + gn] = v;
    if (Ch) Ch[gm * (size_t)ldc + gn] = (_Float16)v;
  }
}

// ---------------------------------------------------------------------------
// Generic WMMA GEMM: C[M,N] = act(A[M,K] * B[K,N] + bias) (+ Res)
// A: f16 row-major (stride lda). Bt = B transposed: [N][ldb] f16 row-major.
// Block tile 64x64, 8 waves (4x2), each wave a 16x32 strip (2 WMMA / K-step).
// Double-buffered LDS staged via global_load_async_to_lds_b128; one
// s_wait_asynccnt + barrier per 32-wide K-step.
// Requires: M % 64 == 0, K % 32 == 0.
// ---------------------------------------------------------------------------
__global__ __launch_bounds__(256) void k_gemm(
    const _Float16* __restrict__ A, int lda,
    const _Float16* __restrict__ Bt, int ldb,
    const float* __restrict__ bias,
    float* __restrict__ C, _Float16* __restrict__ Ch, int ldc,
    const float* __restrict__ Res, int ldr,
    int M, int N, int K, int act)
{
  __shared__ _Float16 As[2][64 * 40];
  __shared__ _Float16 Bs[2][64 * 40];
  const int tid  = threadIdx.x;
  const int m0   = blockIdx.y * 64;
  const int n0   = blockIdx.x * 64;
  const int wave = tid >> 5, lane = tid & 31;
  const int wm = wave >> 1, wn = wave & 1;
  const int fr = lane & 15, hi = lane >> 4;
  const int srow = tid >> 2, sseg = (tid & 3) << 3;

  v8f acc0, acc1;
  for (int j = 0; j < 8; ++j) { acc0[j] = 0.f; acc1[j] = 0.f; }

  const _Float16* Ap = A + (size_t)(m0 + srow) * lda + sseg;
  const bool bval = (n0 + srow) < N;
  const _Float16* Bp = Bt + (size_t)(n0 + srow) * ldb + sseg;

  const unsigned ldsA[2] = { lds_off(As[0] + srow * 40 + sseg),
                             lds_off(As[1] + srow * 40 + sseg) };
  const unsigned ldsB[2] = { lds_off(Bs[0] + srow * 40 + sseg),
                             lds_off(Bs[1] + srow * 40 + sseg) };

  auto stage = [&](int ks, int buf) {
    async_ld_b128(Ap + ks * 32, ldsA[buf]);
    if (bval) {
      async_ld_b128(Bp + ks * 32, ldsB[buf]);
    } else {
      v8h z; zero8(z);
      *(v8h*)(Bs[buf] + srow * 40 + sseg) = z;
    }
  };

  stage(0, 0);
  wait_async0();
  __syncthreads();

  const int nk = K >> 5;
  for (int ks = 0; ks < nk; ++ks) {
    const int cur = ks & 1, nxt = cur ^ 1;
    if (ks + 1 < nk) {
      stage(ks + 1, nxt);
      if (ks + 2 < nk) __builtin_prefetch(Ap + (ks + 2) * 32, 0, 1);
    }
    union { v16h v; v8h h[2]; } af, b0, b1;
    const _Float16* ar = As[cur] + (wm * 16 + fr) * 40;
    af.h[0] = *(const v8h*)(ar + hi * 16);
    af.h[1] = *(const v8h*)(ar + hi * 16 + 8);
    const _Float16* br0 = Bs[cur] + (wn * 32 + fr) * 40 + hi * 16;
    b0.h[0] = *(const v8h*)(br0);
    b0.h[1] = *(const v8h*)(br0 + 8);
    const _Float16* br1 = Bs[cur] + (wn * 32 + 16 + fr) * 40 + hi * 16;
    b1.h[0] = *(const v8h*)(br1);
    b1.h[1] = *(const v8h*)(br1 + 8);
    acc0 = __builtin_amdgcn_wmma_f32_16x16x32_f16(false, af.v, false, b0.v,
                                                  (short)0, acc0, false, false);
    acc1 = __builtin_amdgcn_wmma_f32_16x16x32_f16(false, af.v, false, b1.v,
                                                  (short)0, acc1, false, false);
    wait_async0();
    __syncthreads();
  }
  const int gnA = n0 + wn * 32 + fr;
  const int mbase = m0 + wm * 16 + hi * 8;
  d_epilogue(acc0, gnA,      mbase, N, bias, C, Ch, ldc, Res, ldr, act);
  d_epilogue(acc1, gnA + 16, mbase, N, bias, C, Ch, ldc, Res, ldr, act);
}

// ---------------------------------------------------------------------------
// Implicit-GEMM 3x3 conv, pad == dil, NHWC f16 in (token stride inStride),
// Wt packed f16 [Co][(ky*3+kx)*Cin + ci]. Cin multiple of 32.
// Same 64x64 async double-buffered pipeline; K-stream flattened over taps.
// ---------------------------------------------------------------------------
__global__ __launch_bounds__(256) void k_conv3x3(
    const _Float16* __restrict__ In, int inStride, int Cin,
    const _Float16* __restrict__ Wt,
    const float* __restrict__ bias,
    float* __restrict__ Out, _Float16* __restrict__ OutH, int outStride, int Cout,
    const float* __restrict__ Res, int resStride,
    int Hh, int Ww, int dil, int act)
{
  __shared__ _Float16 As[2][64 * 40];
  __shared__ _Float16 Bs[2][64 * 40];
  const int tid  = threadIdx.x;
  const int m0   = blockIdx.y * 64;
  const int n0   = blockIdx.x * 64;
  const int wave = tid >> 5, lane = tid & 31;
  const int wm = wave >> 1, wn = wave & 1;
  const int fr = lane & 15, hi = lane >> 4;
  const int srow = tid >> 2, sseg = (tid & 3) << 3;

  // staging thread's token coordinates
  const int mt = m0 + srow;
  const int bb = mt / (Hh * Ww);
  const int yy = (mt / Ww) % Hh;
  const int xx = mt % Ww;
  const bool bval = (n0 + srow) < Cout;
  const _Float16* Bp = Wt + (size_t)(n0 + srow) * (9 * Cin) + sseg;

  const unsigned ldsA[2] = { lds_off(As[0] + srow * 40 + sseg),
                             lds_off(As[1] + srow * 40 + sseg) };
  const unsigned ldsB[2] = { lds_off(Bs[0] + srow * 40 + sseg),
                             lds_off(Bs[1] + srow * 40 + sseg) };

  v8f acc0, acc1;
  for (int j = 0; j < 8; ++j) { acc0[j] = 0.f; acc1[j] = 0.f; }

  const int ksteps = Cin >> 5;
  const int nkk = 9 * ksteps;

  auto stageA = [&](int kk, int buf) {
    const int tap = kk / ksteps;
    const int kc  = kk - tap * ksteps;
    const int ky = tap / 3, kx = tap % 3;
    const int oy = yy + (ky - 1) * dil;
    const int ox = xx + (kx - 1) * dil;
    const bool valid = (oy >= 0) && (oy < Hh) && (ox >= 0) && (ox < Ww);
    if (valid) {
      async_ld_b128(In + ((size_t)(bb * Hh + oy) * Ww + ox) * inStride +
                        (kc << 5) + sseg,
                    ldsA[buf]);
    } else {
      v8h z; zero8(z);
      *(v8h*)(As[buf] + srow * 40 + sseg) = z;
    }
  };
  auto stageB = [&](int kk, int buf) {
    // weight K index is linear: tap*Cin + kc*32 == kk*32
    if (bval) {
      async_ld_b128(Bp + kk * 32, ldsB[buf]);
    } else {
      v8h z; zero8(z);
      *(v8h*)(Bs[buf] + srow * 40 + sseg) = z;
    }
  };

  stageA(0, 0);
  stageB(0, 0);
  wait_async0();
  __syncthreads();

  for (int kk = 0; kk < nkk; ++kk) {
    const int cur = kk & 1, nxt = cur ^ 1;
    if (kk + 1 < nkk) {
      stageA(kk + 1, nxt);
      stageB(kk + 1, nxt);
      if (kk + 2 < nkk) __builtin_prefetch(Bp + (kk + 2) * 32, 0, 1);
    }
    union { v16h v; v8h h[2]; } af, b0, b1;
    const _Float16* ar = As[cur] + (wm * 16 + fr) * 40;
    af.h[0] = *(const v8h*)(ar + hi * 16);
    af.h[1] = *(const v8h*)(ar + hi * 16 + 8);
    const _Float16* br0 = Bs[cur] + (wn * 32 + fr) * 40 + hi * 16;
    b0.h[0] = *(const v8h*)(br0);
    b0.h[1] = *(const v8h*)(br0 + 8);
    const _Float16* br1 = Bs[cur] + (wn * 32 + 16 + fr) * 40 + hi * 16;
    b1.h[0] = *(const v8h*)(br1);
    b1.h[1] = *(const v8h*)(br1 + 8);
    acc0 = __builtin_amdgcn_wmma_f32_16x16x32_f16(false, af.v, false, b0.v,
                                                  (short)0, acc0, false, false);
    acc1 = __builtin_amdgcn_wmma_f32_16x16x32_f16(false, af.v, false, b1.v,
                                                  (short)0, acc1, false, false);
    wait_async0();
    __syncthreads();
  }
  const int gnA = n0 + wn * 32 + fr;
  const int mbase = m0 + wm * 16 + hi * 8;
  d_epilogue(acc0, gnA,      mbase, Cout, bias, Out, OutH, outStride, Res, resStride, act);
  d_epilogue(acc1, gnA + 16, mbase, Cout, bias, Out, OutH, outStride, Res, resStride, act);
}

// ----------------------------- helper kernels ------------------------------
__global__ void k_in_tr(const float* __restrict__ in, float* __restrict__ X,
                        int Bn, int Hh, int Ww, int C) {
  int gid = blockIdx.x * 256 + threadIdx.x;
  int total = Bn * C * Hh * Ww;
  if (gid >= total) return;
  int x = gid % Ww, y = (gid / Ww) % Hh, c = (gid / (Ww * Hh)) % C, b = gid / (Ww * Hh * C);
  X[(((size_t)b * Hh + y) * Ww + x) * C + c] = in[gid];
}

__global__ void k_out_tr(const float* __restrict__ X, float* __restrict__ out,
                         int Bn, int Hh, int Ww, int C) {
  int gid = blockIdx.x * 256 + threadIdx.x;
  int total = Bn * C * Hh * Ww;
  if (gid >= total) return;
  int x = gid % Ww, y = (gid / Ww) % Hh, c = (gid / (Ww * Hh)) % C, b = gid / (Ww * Hh * C);
  out[gid] = X[(((size_t)b * Hh + y) * Ww + x) * C + c];
}

__global__ void k_ln(const float* __restrict__ X, const float* __restrict__ g,
                     const float* __restrict__ bta, _Float16* __restrict__ Y,
                     int Mtot, int C) {
  int m = blockIdx.x * 256 + threadIdx.x;
  if (m >= Mtot) return;
  const float* xr = X + (size_t)m * C;
  float mean = 0.f;
  for (int c = 0; c < C; ++c) mean += xr[c];
  mean /= (float)C;
  float var = 0.f;
  for (int c = 0; c < C; ++c) { float d = xr[c] - mean; var += d * d; }
  var /= (float)C;
  float inv = rsqrtf(var + 1e-5f);
  _Float16* yr = Y + (size_t)m * C;
  for (int c = 0; c < C; ++c) yr[c] = (_Float16)((xr[c] - mean) * inv * g[c] + bta[c]);
}

__global__ void k_pack_t(const float* __restrict__ W, _Float16* __restrict__ Bt, int K, int N) {
  int i = blockIdx.x * 256 + threadIdx.x;
  if (i >= K * N) return;
  int k = i / N, n = i % N;
  Bt[(size_t)n * K + k] = (_Float16)W[i];
}

__global__ void k_pack_conv(const float* __restrict__ W, _Float16* __restrict__ Bt, int Co, int Ci) {
  int i = blockIdx.x * 256 + threadIdx.x;
  if (i >= Co * Ci * 9) return;
  int kx = i % 3, ky = (i / 3) % 3, ci = (i / 9) % Ci, co = i / (9 * Ci);
  Bt[(size_t)co * (9 * Ci) + (ky * 3 + kx) * Ci + ci] = (_Float16)W[i];
}

__global__ void k_pack_dsw(const float* __restrict__ W, _Float16* __restrict__ Bt, int O, int C) {
  int i = blockIdx.x * 256 + threadIdx.x;
  if (i >= O * C * 9) return;
  int kk = i % 9, c = (i / 9) % C, o = i / (9 * C);
  Bt[(size_t)o * (9 * C) + kk * C + c] = (_Float16)W[i];
}

__global__ void k_moe_gate(const _Float16* __restrict__ Xh, int ldx,
                           const float* __restrict__ gW, const float* __restrict__ gb,
                           int Mtot, int C, int E, int* __restrict__ sel,
                           float* __restrict__ gate) {
  int m = blockIdx.x * 256 + threadIdx.x;
  if (m >= Mtot) return;
  float l[8];
  for (int e = 0; e < E; ++e) l[e] = gb[e];
  const _Float16* xr = Xh + (size_t)m * ldx;
  for (int c = 0; c < C; ++c) {
    float xv = (float)xr[c];
    for (int e = 0; e < E; ++e) l[e] += xv * gW[c * E + e];
  }
  int arg = 0; float lmax = l[0];
  for (int e = 1; e < E; ++e) if (l[e] > lmax) { lmax = l[e]; arg = e; }
  float s = 0.f;
  for (int e = 0; e < E; ++e) s += expf(l[e] - lmax);
  sel[m] = arg;
  gate[m] = 1.f / s;  // == max of softmax
}

__global__ void k_dmoe_apply(float* __restrict__ X, const _Float16* __restrict__ Ye,
                             const int* __restrict__ sel, const float* __restrict__ gate,
                             int Mtot, int C) {
  int gid = blockIdx.x * 256 + threadIdx.x;
  if (gid >= Mtot * C) return;
  int m = gid / C, c = gid % C;
  float h = (float)Ye[((size_t)sel[m] * Mtot + m) * C + c] * gate[m];
  X[(size_t)m * C + c] += (h >= 0.f) ? h : 0.01f * h;
}

__global__ void k_amoe_select(_Float16* __restrict__ outh, const _Float16* __restrict__ Ye,
                              const int* __restrict__ sel, const float* __restrict__ gate,
                              int Mtot, int C) {
  int gid = blockIdx.x * 256 + threadIdx.x;
  if (gid >= Mtot * C) return;
  int m = gid / C, c = gid % C;
  outh[(size_t)m * C + c] =
      (_Float16)((float)Ye[((size_t)sel[m] * Mtot + m) * C + c] * gate[m]);
}

// Bilinear deformable sampling: S[m][k*64+c]
__global__ void k_dsample(const float* __restrict__ Xf, int ldx,
                          const float* __restrict__ Off, _Float16* __restrict__ S,
                          int Hh, int Ww, int morph, int dil, int Mtot) {
  int gid = blockIdx.x * 256 + threadIdx.x;
  if (gid >= Mtot * 9 * 64) return;
  int c = gid & 63;
  int k = (gid >> 6) % 9;
  int m = gid / (9 * 64);
  int b = m / (Hh * Ww), y = (m / Ww) % Hh, x = m % Ww;
  float dy = Off[(size_t)m * 18 + k];
  float dx = Off[(size_t)m * 18 + 9 + k];
  float ks = (float)(k - 4) * (float)dil;
  float ys, xs;
  if (morph == 0) { ys = (float)y + dy;      xs = (float)x + ks + dx; }
  else            { ys = (float)y + ks + dy; xs = (float)x + dx; }
  ys = fminf(fmaxf(ys, 0.f), (float)(Hh - 1));
  xs = fminf(fmaxf(xs, 0.f), (float)(Ww - 1));
  float y0f = floorf(ys), x0f = floorf(xs);
  float wy = ys - y0f, wx = xs - x0f;
  int y0 = (int)y0f, x0 = (int)x0f;
  int y1 = (y0 + 1 < Hh) ? y0 + 1 : Hh - 1;
  int x1 = (x0 + 1 < Ww) ? x0 + 1 : Ww - 1;
  size_t base = (size_t)b * Hh * Ww;
  const float v00 = Xf[(base + (size_t)y0 * Ww + x0) * ldx + c];
  const float v01 = Xf[(base + (size_t)y0 * Ww + x1) * ldx + c];
  const float v10 = Xf[(base + (size_t)y1 * Ww + x0) * ldx + c];
  const float v11 = Xf[(base + (size_t)y1 * Ww + x1) * ldx + c];
  float v = v00 * (1.f - wy) * (1.f - wx) + v01 * (1.f - wy) * wx +
            v10 * wy * (1.f - wx) + v11 * wy * wx;
  S[(size_t)m * 576 + k * 64 + c] = (_Float16)v;
}

// Row attention: seq=64 (width), dim=64, per (b, y) row; output written transposed.
__global__ __launch_bounds__(64) void k_attn(const _Float16* __restrict__ qkv,
                                             _Float16* __restrict__ outh,
                                             int Hh, int Ww, float scale) {
  __shared__ float qs[64][64];
  __shared__ float ks_[64][64];
  __shared__ float vs_[64][64];
  const int b = blockIdx.x / Hh;
  const int y = blockIdx.x % Hh;
  const int x = threadIdx.x;
  const size_t m = ((size_t)b * Hh + y) * Ww + x;
  for (int c = 0; c < 64; ++c) {
    qs[x][c]  = (float)qkv[m * 192 + c];
    ks_[x][c] = (float)qkv[m * 192 + 64 + c];
    vs_[x][c] = (float)qkv[m * 192 + 128 + c];
  }
  __syncthreads();
  float sc[64];
  float mx = -1e30f;
  for (int u = 0; u < 64; ++u) {
    float d = 0.f;
    for (int c = 0; c < 64; ++c) d += qs[x][c] * ks_[u][c];
    d *= scale;
    sc[u] = d;
    mx = fmaxf(mx, d);
  }
  float s = 0.f;
  for (int u = 0; u < 64; ++u) { sc[u] = expf(sc[u] - mx); s += sc[u]; }
  const float inv = 1.f / s;
  const size_t mo = ((size_t)b * Hh + x) * Ww + y;  // baked-in transpose
  for (int c = 0; c < 64; ++c) {
    float o = 0.f;
    for (int u = 0; u < 64; ++u) o += sc[u] * vs_[u][c];
    outh[mo * 64 + c] = (_Float16)(o * inv);
  }
}

// ---------------------------------------------------------------------------
extern "C" void kernel_launch(void* const* d_in, const int* in_sizes, int n_in,
                              void* d_out, int out_size, void* d_ws, size_t ws_size,
                              hipStream_t stream) {
  (void)in_sizes; (void)n_in; (void)out_size; (void)ws_size;
  const int Bn = 8, Hh = 64, Ww = 64, DIMc = 192, HDc = 64, E = 5;
  const int M = Bn * Hh * Ww;

  const float* x_in  = (const float*)d_in[0];
  const float* n1_g = (const float*)d_in[1];  const float* n1_b = (const float*)d_in[2];
  const float* n2_g = (const float*)d_in[3];  const float* n2_b = (const float*)d_in[4];
  const float* n3_g = (const float*)d_in[5];  const float* n3_b = (const float*)d_in[6];
  const float* n4_g = (const float*)d_in[7];  const float* n4_b = (const float*)d_in[8];
  const float* dec_W = (const float*)d_in[9];  const float* dec_b = (const float*)d_in[10];
  const float* dmoe_gW = (const float*)d_in[11]; const float* dmoe_gb = (const float*)d_in[12];
  const float* dmoe_eW = (const float*)d_in[13]; const float* dmoe_eb = (const float*)d_in[14];
  const float* p1_W = (const float*)d_in[15]; const float* p1_b = (const float*)d_in[16];
  const float* p3_W = (const float*)d_in[17]; const float* p3_b = (const float*)d_in[18];
  const float* e1_W = (const float*)d_in[19]; const float* e1_b = (const float*)d_in[20];
  const float* e2_W = (const float*)d_in[21]; const float* e2_b = (const float*)d_in[22];
  const float* ds_oW = (const float*)d_in[23]; const float* ds_ob = (const float*)d_in[24];
  const float* ds_Wp = (const float*)d_in[25]; const float* ds_bp = (const float*)d_in[26];
  const float* lin_W = (const float*)d_in[27]; const float* lin_b = (const float*)d_in[28];
  const float* amoe_gW = (const float*)d_in[29]; const float* amoe_gb = (const float*)d_in[30];
  const float* amoe_eW = (const float*)d_in[31]; const float* amoe_eb = (const float*)d_in[32];
  const float* qkv_W = (const float*)d_in[33];
  const float* att_pW = (const float*)d_in[34]; const float* att_pb = (const float*)d_in[35];
  const float* u1_W = (const float*)d_in[36]; const float* u1_b = (const float*)d_in[37];
  const float* u2_W = (const float*)d_in[38]; const float* u2_b = (const float*)d_in[39];
  const float* u3_W = (const float*)d_in[40]; const float* u3_b = (const float*)d_in[41];
  const float* c1_W = (const float*)d_in[42]; const float* c1_b = (const float*)d_in[43];
  const float* fc1_W = (const float*)d_in[44]; const float* fc1_b = (const float*)d_in[45];
  const float* fc2_W = (const float*)d_in[46]; const float* fc2_b = (const float*)d_in[47];
  float* outp = (float*)d_out;

  char* wsb = (char*)d_ws;
  size_t cur = 0;
  auto alloc = [&](size_t bytes) { size_t r = cur; cur += (bytes + 255) & ~(size_t)255; return r; };
  float*    X     = (float*)   (wsb + alloc((size_t)M * DIMc * 4));  // running x (NHWC f32)
  float*    XN    = (float*)   (wsb + alloc((size_t)M * DIMc * 4));  // p1 output f32
  float*    OFFb  = (float*)   (wsb + alloc((size_t)M * 18 * 4));    // offset-conv out
  _Float16* HA    = (_Float16*)(wsb + alloc((size_t)M * 768 * 2));   // S / hcat / fc-mid
  _Float16* HLN   = (_Float16*)(wsb + alloc((size_t)M * DIMc * 2));  // LN outputs
  _Float16* XNH   = (_Float16*)(wsb + alloc((size_t)M * DIMc * 2));  // p1 output f16
  _Float16* HC    = (_Float16*)(wsb + alloc((size_t)M * DIMc * 2));  // dec / branchcat / qkv
  _Float16* HXM   = (_Float16*)(wsb + alloc((size_t)M * HDc * 2));
  _Float16* HXM2  = (_Float16*)(wsb + alloc((size_t)M * HDc * 2));
  _Float16* HATT  = (_Float16*)(wsb + alloc((size_t)M * HDc * 2));
  _Float16* OUTSb = (_Float16*)(wsb + alloc((size_t)M * DIMc * 2));
  _Float16* YE    = (_Float16*)(wsb + alloc((size_t)E * M * DIMc * 2));
  int*      SEL   = (int*)     (wsb + alloc((size_t)M * 4));
  float*    GATE  = (float*)   (wsb + alloc((size_t)M * 4));
  _Float16* PK    = (_Float16*)(wsb + alloc((size_t)2 * 1024 * 1024));

  auto cdiv = [](int a, int b) { return (a + b - 1) / b; };
  const dim3 blk256(256);

  auto packT = [&](const float* Wp, int K, int N) {
    k_pack_t<<<cdiv(K * N, 256), blk256, 0, stream>>>(Wp, PK, K, N);
  };
  auto packC = [&](const float* Wp, int Co, int Ci) {
    k_pack_conv<<<cdiv(Co * Ci * 9, 256), blk256, 0, stream>>>(Wp, PK, Co, Ci);
  };
  auto gemm = [&](const _Float16* A, int lda, const float* bias,
                  float* C, _Float16* Ch, int ldc, const float* Res, int ldr,
                  int N, int K, int act) {
    dim3 g(cdiv(N, 64), M / 64);
    k_gemm<<<g, blk256, 0, stream>>>(A, lda, PK, K, bias, C, Ch, ldc, Res, ldr, M, N, K, act);
  };
  auto conv = [&](const _Float16* In, int inStride, int Cin, const float* bias,
                  float* Out, _Float16* OutH, int ldo, int Cout,
                  const float* Res, int ldr, int dil, int act) {
    dim3 g(cdiv(Cout, 64), M / 64);
    k_conv3x3<<<g, blk256, 0, stream>>>(In, inStride, Cin, PK, bias, Out, OutH, ldo, Cout,
                                        Res, ldr, Hh, Ww, dil, act);
  };

  // ---- NCHW -> NHWC ----
  k_in_tr<<<cdiv(M * DIMc, 256), blk256, 0, stream>>>(x_in, X, Bn, Hh, Ww, DIMc);

  // ---- decoder conv + sparse MoE + leaky residual ----
  k_ln<<<cdiv(M, 256), blk256, 0, stream>>>(X, n4_g, n4_b, HLN, M, DIMc);
  packC(dec_W, DIMc, DIMc);
  conv(HLN, DIMc, DIMc, dec_b, nullptr, HC, DIMc, DIMc, nullptr, 0, 1, ACT_NONE);
  k_moe_gate<<<cdiv(M, 256), blk256, 0, stream>>>(HC, DIMc, dmoe_gW, dmoe_gb, M, DIMc, E, SEL, GATE);
  for (int e = 0; e < E; ++e) {
    packT(dmoe_eW + (size_t)e * DIMc * DIMc, DIMc, DIMc);
    gemm(HC, DIMc, dmoe_eb + e * DIMc, nullptr, YE + (size_t)e * M * DIMc, DIMc,
         nullptr, 0, DIMc, DIMc, ACT_NONE);
  }
  k_dmoe_apply<<<cdiv(M * DIMc, 256), blk256, 0, stream>>>(X, YE, SEL, GATE, M, DIMc);

  // ---- p1 projection ----
  k_ln<<<cdiv(M, 256), blk256, 0, stream>>>(X, n1_g, n1_b, HLN, M, DIMc);
  packT(p1_W, DIMc, DIMc);
  gemm(HLN, DIMc, p1_b, XN, XNH, DIMc, nullptr, 0, DIMc, DIMc, ACT_NONE);

  // ---- 3 branches ----
  for (int i = 0; i < 3; ++i) {
    if (i == 0) {
      packC(e1_W, HDc, HDc);
      conv(XNH, DIMc, HDc, e1_b, nullptr, HC, 2 * HDc, HDc, nullptr, 0, 1, ACT_NONE);
      packC(e2_W, HDc, HDc);
      conv(XNH, DIMc, HDc, e2_b, nullptr, HC + HDc, 2 * HDc, HDc, nullptr, 0, 2, ACT_NONE);
    } else {
      const int morph = (i == 2) ? 1 : 0;
      for (int t = 0; t < 2; ++t) {
        const int j = (i - 1) * 2 + t;
        const int dl = 1 + t;
        packC(ds_oW + (size_t)j * 18 * HDc * 9, 18, HDc);
        conv(XNH + i * HDc, DIMc, HDc, ds_ob + j * 18, OFFb, nullptr, 18, 18,
             nullptr, 0, 1, ACT_TANH);
        k_dsample<<<cdiv(M * 9 * HDc, 256), blk256, 0, stream>>>(
            XN + i * HDc, DIMc, OFFb, HA, Hh, Ww, morph, dl, M);
        k_pack_dsw<<<cdiv(HDc * HDc * 9, 256), blk256, 0, stream>>>(
            ds_Wp + (size_t)j * HDc * HDc * 9, PK, HDc, HDc);
        gemm(HA, 9 * HDc, ds_bp + j * HDc, nullptr, HC + t * HDc, 2 * HDc,
             nullptr, 0, HDc, 9 * HDc, ACT_RELU);
      }
    }
    packT(lin_W + (size_t)i * 2 * HDc * HDc, 2 * HDc, HDc);
    gemm(HC, 2 * HDc, lin_b + i * HDc, nullptr, HXM, HDc, nullptr, 0, HDc, 2 * HDc, ACT_NONE);
    k_moe_gate<<<cdiv(M, 256), blk256, 0, stream>>>(HXM, HDc, amoe_gW + i * HDc * E,
                                                    amoe_gb + i * E, M, HDc, E, SEL, GATE);
    for (int e = 0; e < E; ++e) {
      packT(amoe_eW + (size_t)(i * E + e) * HDc * HDc, HDc, HDc);
      gemm(HXM, HDc, amoe_eb + (i * E + e) * HDc, nullptr, YE + (size_t)e * M * HDc, HDc,
           nullptr, 0, HDc, HDc, ACT_NONE);
    }
    k_amoe_select<<<cdiv(M * HDc, 256), blk256, 0, stream>>>(HXM2, YE, SEL, GATE, M, HDc);
    packT(qkv_W + (size_t)i * HDc * 3 * HDc, HDc, 3 * HDc);
    gemm(HXM2, HDc, nullptr, nullptr, HC, DIMc, nullptr, 0, 3 * HDc, HDc, ACT_NONE);
    k_attn<<<Bn * Hh, 64, 0, stream>>>(HC, HATT, Hh, Ww, 0.35355339059327373f);
    packT(att_pW + (size_t)i * HDc * HDc, HDc, HDc);
    gemm(HATT, HDc, att_pb + i * HDc, nullptr, OUTSb + i * HDc, DIMc,
         nullptr, 0, HDc, HDc, ACT_NONE);
  }

  // ---- p3 + residual ----
  packT(p3_W, DIMc, DIMc);
  gemm(OUTSb, DIMc, p3_b, X, nullptr, DIMc, X, DIMc, DIMc, DIMc, ACT_NONE);

  // ---- multi-dilation conv MLP ----
  k_ln<<<cdiv(M, 256), blk256, 0, stream>>>(X, n2_g, n2_b, HLN, M, DIMc);
  packC(u1_W, DIMc, DIMc);
  conv(HLN, DIMc, DIMc, u1_b, nullptr, HA, 3 * DIMc, DIMc, nullptr, 0, 1, ACT_GELU);
  packC(u2_W, DIMc, DIMc);
  conv(HLN, DIMc, DIMc, u2_b, nullptr, HA + DIMc, 3 * DIMc, DIMc, nullptr, 0, 2, ACT_GELU);
  packC(u3_W, DIMc, DIMc);
  conv(HLN, DIMc, DIMc, u3_b, nullptr, HA + 2 * DIMc, 3 * DIMc, DIMc, nullptr, 0, 3, ACT_GELU);
  packC(c1_W, DIMc, 3 * DIMc);
  conv(HA, 3 * DIMc, 3 * DIMc, c1_b, X, nullptr, DIMc, DIMc, X, DIMc, 1, ACT_GELU);

  // ---- MLP ----
  k_ln<<<cdiv(M, 256), blk256, 0, stream>>>(X, n3_g, n3_b, HLN, M, DIMc);
  packT(fc1_W, DIMc, 4 * DIMc);
  gemm(HLN, DIMc, fc1_b, nullptr, HA, 4 * DIMc, nullptr, 0, 4 * DIMc, DIMc, ACT_GELU);
  packT(fc2_W, 4 * DIMc, DIMc);
  gemm(HA, 4 * DIMc, fc2_b, X, nullptr, DIMc, X, DIMc, DIMc, 4 * DIMc, ACT_NONE);

  // ---- NHWC -> NCHW ----
  k_out_tr<<<cdiv(M * DIMc, 256), blk256, 0, stream>>>(X, outp, Bn, Hh, Ww, DIMc);
}
